// SuperconGNN_75668733821136
// MI455X (gfx1250) — compile-verified, matrix operand order checked by reading, not run on hardware
//
#include <hip/hip_runtime.h>
#include <math.h>

typedef __attribute__((ext_vector_type(16))) _Float16 v16h;
typedef __attribute__((ext_vector_type(8)))  _Float16 v8h;
typedef __attribute__((ext_vector_type(8)))  float    v8f;

union F16x16 { v16h v; v8h h[2]; };

#define HPAD   256      // padded node-feature width (max irrep dim = 256)
#define NSC    48
#define DISTD  32
#define EATTRW (2 + DISTD)   // 34
#define EATTRP 64            // K-padded
#define NEF    144
#define NEFP   160           // K-padded (multiple of 32)
#define CHUNK  512

// ------------------------------------------------------------------
// Tensor-product path metadata (built on host, passed via kernargs)
// ------------------------------------------------------------------
struct TpPath { int in_off, m1, l1, sh_off, l2, out_off, m3, l3, woff; float coeff; };
struct TpMeta { int np; TpPath p[32]; };
struct IrrList { int n; int m[6]; int l[6]; int p[6]; };

// ------------------------------------------------------------------
// Wigner 3j (real basis, Frobenius-normalized) computed on device
// ------------------------------------------------------------------
__device__ double dfact_(int n) { double r = 1.0; for (int i = 2; i <= n; ++i) r *= (double)i; return r; }
__device__ double tri_f_(int a, int b, int c) {
    return dfact_(a + b - c) * dfact_(a - b + c) * dfact_(-a + b + c) / dfact_(a + b + c + 1);
}
__device__ double cg_(int j1, int m1, int j2, int m2, int j3, int m3) {
    if (m1 + m2 != m3) return 0.0;
    double pref = sqrt((double)(2 * j3 + 1) * tri_f_(j1, j2, j3) *
                       dfact_(j3 + m3) * dfact_(j3 - m3) * dfact_(j1 - m1) *
                       dfact_(j1 + m1) * dfact_(j2 - m2) * dfact_(j2 + m2));
    double s = 0.0;
    for (int k = 0; k <= j1 + j2 - j3; ++k) {
        int d1 = j1 + j2 - j3 - k, d2 = j1 - m1 - k, d3 = j2 + m2 - k;
        int d4 = j3 - j2 + m1 + k, d5 = j3 - j1 - m2 + k;
        if (d1 < 0 || d2 < 0 || d3 < 0 || d4 < 0 || d5 < 0) continue;
        double den = dfact_(k) * dfact_(d1) * dfact_(d2) * dfact_(d3) * dfact_(d4) * dfact_(d5);
        s += ((k & 1) ? -1.0 : 1.0) / den;
    }
    return pref * s;
}
__device__ void qmat_(int l, double qr[5][5], double qi[5][5]) {
    for (int a = 0; a < 5; ++a) for (int b = 0; b < 5; ++b) { qr[a][b] = 0.0; qi[a][b] = 0.0; }
    const double is2 = 0.70710678118654752440;
    for (int m = -l; m < 0; ++m) { qr[l + m][l - m] = is2; qi[l + m][l + m] = -is2; }
    qr[l][l] = 1.0;
    for (int m = 1; m <= l; ++m) {
        double sg = (m & 1) ? -1.0 : 1.0;
        qr[l + m][l + m] = sg * is2; qi[l + m][l - m] = sg * is2;
    }
    if (l == 1) {          // multiply by (-i)
        for (int a = 0; a < 5; ++a) for (int b = 0; b < 5; ++b) {
            double r = qr[a][b], im = qi[a][b]; qr[a][b] = im; qi[a][b] = -r;
        }
    } else if (l == 2) {   // multiply by (-i)^2 = -1
        for (int a = 0; a < 5; ++a) for (int b = 0; b < 5; ++b) { qr[a][b] = -qr[a][b]; qi[a][b] = -qi[a][b]; }
    }
}

__global__ void k_w3j(float* __restrict__ tab) {
    int t = threadIdx.x;
    if (t >= 27) return;
    int l1 = t / 9, l2 = (t / 3) % 3, l3 = t % 3;
    int lo = l1 - l2; if (lo < 0) lo = -lo;
    if (l3 < lo || l3 > l1 + l2) return;
    int d1 = 2 * l1 + 1, d2 = 2 * l2 + 1, d3 = 2 * l3 + 1;
    double q1r[5][5], q1i[5][5], q2r[5][5], q2i[5][5], q3r[5][5], q3i[5][5];
    qmat_(l1, q1r, q1i); qmat_(l2, q2r, q2i); qmat_(l3, q3r, q3i);
    float* o = tab + t * 125;
    double nrm = 0.0;
    for (int j = 0; j < d1; ++j)
      for (int ll = 0; ll < d2; ++ll)
        for (int n = 0; n < d3; ++n) {
            double acc = 0.0;
            for (int i = 0; i < d1; ++i)
              for (int k = 0; k < d2; ++k) {
                  int m = (i - l1) + (k - l2) + l3;
                  if (m < 0 || m >= d3) continue;
                  double C = cg_(l1, i - l1, l2, k - l2, l3, m - l3);
                  if (C == 0.0) continue;
                  double ar = q1r[i][j],  ai = q1i[i][j];
                  double br = q2r[k][ll], bi = q2i[k][ll];
                  double cr = q3r[m][n],  ci = -q3i[m][n];
                  double abr = ar * br - ai * bi, abi = ar * bi + ai * br;
                  acc += (abr * cr - abi * ci) * C;
              }
            o[(j * d2 + ll) * d3 + n] = (float)acc;
            nrm += acc * acc;
        }
    float inv = (float)(1.0 / sqrt(nrm));
    for (int x = 0; x < d1 * d2 * d3; ++x) o[x] *= inv;
}

// ------------------------------------------------------------------
// Utility kernels
// ------------------------------------------------------------------
__global__ void k_zero(float* p, int n) {
    int i = blockIdx.x * blockDim.x + threadIdx.x;
    if (i < n) p[i] = 0.0f;
}

__global__ void k_cnt(const int* __restrict__ src, float* __restrict__ cnt, int E) {
    int e = blockIdx.x * blockDim.x + threadIdx.x;
    if (e < E) atomicAdd(&cnt[src[e]], 1.0f);
}

// Weight convert + transpose: B[K,N] (f32) -> BT[Nalloc, Kp] (f16, zero padded)
__global__ void k_cvtBT(const float* __restrict__ B, int K, int N,
                        _Float16* __restrict__ BT, int Kp, int Nalloc) {
    int i = blockIdx.x * blockDim.x + threadIdx.x;
    if (i >= Nalloc * Kp) return;
    int n = i / Kp, k = i % Kp;
    float v = (n < N && k < K) ? B[(size_t)k * N + n] : 0.0f;
    BT[i] = (_Float16)v;
}

// ------------------------------------------------------------------
// Node embedding: h[:,0:48] = emb_table[x_cat] + x_scalar @ node_W + b
// ------------------------------------------------------------------
__global__ void k_node_init(const int* __restrict__ xcat, const float* __restrict__ xsc,
                            const float* __restrict__ emb, const float* __restrict__ nW,
                            const float* __restrict__ nb, float* __restrict__ h, int N) {
    int i = blockIdx.x * blockDim.x + threadIdx.x;
    if (i >= N * HPAD) return;
    int n = i / HPAD, c = i % HPAD;
    float v = 0.0f;
    if (c < NSC) {
        v = emb[xcat[n] * NSC + c] + nb[c];
        for (int k = 0; k < 6; ++k) v += xsc[n * 6 + k] * nW[k * NSC + c];
    }
    h[i] = v;
}

// ------------------------------------------------------------------
// Edge features -> eattr16 (f16, K-padded to 64) + spherical harmonics (f32)
// ------------------------------------------------------------------
__global__ void k_edge_feat(const float* __restrict__ pos, const int* __restrict__ src,
                            const int* __restrict__ dst, const float* __restrict__ eain,
                            _Float16* __restrict__ eattr16, float* __restrict__ esh, int E) {
    int e = blockIdx.x * blockDim.x + threadIdx.x;
    if (e >= E) return;
    int s = src[e], d = dst[e];
    float ex = pos[d * 3 + 0] - pos[s * 3 + 0];
    float ey = pos[d * 3 + 1] - pos[s * 3 + 1];
    float ez = pos[d * 3 + 2] - pos[s * 3 + 2];
    float dist = sqrtf(ex * ex + ey * ey + ez * ez);
    _Float16* a = eattr16 + (size_t)e * EATTRP;
    a[0] = (_Float16)eain[e * 2 + 0];
    a[1] = (_Float16)eain[e * 2 + 1];
    const float step = 5.0f / 31.0f;
    const float gcoef = -0.5f / (step * step);
    for (int i = 0; i < DISTD; ++i) {
        float t = dist - step * (float)i;
        a[2 + i] = (_Float16)__expf(gcoef * t * t);
    }
    for (int i = EATTRW; i < EATTRP; ++i) a[i] = (_Float16)0.0f;
    float inv = 1.0f / dist;
    float x = ex * inv, y = ey * inv, z = ez * inv;
    const float s3 = 1.7320508075688772f, s5 = 2.2360679774997896f, s15 = 3.872983346207417f;
    float* o = esh + e * 9;
    o[0] = 1.0f;        o[1] = s3 * y;      o[2] = s3 * z;     o[3] = s3 * x;
    o[4] = s15 * x * y; o[5] = s15 * y * z; o[6] = 0.5f * s5 * (3.0f * z * z - 1.0f);
    o[7] = s15 * x * z; o[8] = 0.5f * s15 * (x * x - y * y);
}

// ------------------------------------------------------------------
// WMMA GEMM (branch-free hot loop):
//   out = optional_relu(A16[M,Kp] @ BT16[.,Kp]^T + bias)
//   C   (f32, cols < N)       optional
//   C16 (f16, cols < Np16, zero-padded beyond N) optional
// Block = 128 threads = 4 waves; each wave owns a 32x32 tile of a 64x64 block
// tile (2x2 v_wmma_f32_16x16x32_f16 accumulators). Requires M % 64 == 0 and
// all loads stay inside padded buffers, so EXEC is all-ones throughout.
// ------------------------------------------------------------------
__global__ void k_gemm16(const _Float16* __restrict__ A, int lda,
                         const _Float16* __restrict__ BT, int ldb,
                         const float* __restrict__ bias,
                         float* __restrict__ C, int ldc,
                         _Float16* __restrict__ C16, int ldc16,
                         int N, int Kp, int Np16, int relu) {
    int lane = threadIdx.x & 31;
    int wave = threadIdx.x >> 5;                 // 0..3
    int row0 = blockIdx.y * 64 + (wave >> 1) * 32;
    int col0 = blockIdx.x * 64 + (wave & 1) * 32;
    int half = lane >> 4, idx = lane & 15;
    int kb = half * 8;                            // ISA 16-bit fragment K base
    const _Float16* Ar0 = A  + (size_t)(row0 + idx) * lda + kb;
    const _Float16* Ar1 = Ar0 + (size_t)16 * lda;
    const _Float16* Bc0 = BT + (size_t)(col0 + idx) * ldb + kb;
    const _Float16* Bc1 = Bc0 + (size_t)16 * ldb;
    v8f acc00 = {}, acc01 = {}, acc10 = {}, acc11 = {};
    for (int k0 = 0; k0 < Kp; k0 += 32) {
        if (k0 + 32 < Kp) {                       // global_prefetch_b8 next K slab
            __builtin_prefetch(Ar0 + k0 + 32, 0, 0);
            __builtin_prefetch(Bc0 + k0 + 32, 0, 0);
        }
        F16x16 a0, a1, b0, b1;
        a0.h[0] = *(const v8h*)(Ar0 + k0);  a0.h[1] = *(const v8h*)(Ar0 + k0 + 16);
        a1.h[0] = *(const v8h*)(Ar1 + k0);  a1.h[1] = *(const v8h*)(Ar1 + k0 + 16);
        b0.h[0] = *(const v8h*)(Bc0 + k0);  b0.h[1] = *(const v8h*)(Bc0 + k0 + 16);
        b1.h[0] = *(const v8h*)(Bc1 + k0);  b1.h[1] = *(const v8h*)(Bc1 + k0 + 16);
        acc00 = __builtin_amdgcn_wmma_f32_16x16x32_f16(false, a0.v, false, b0.v, (short)0, acc00, false, false);
        acc01 = __builtin_amdgcn_wmma_f32_16x16x32_f16(false, a0.v, false, b1.v, (short)0, acc01, false, false);
        acc10 = __builtin_amdgcn_wmma_f32_16x16x32_f16(false, a1.v, false, b0.v, (short)0, acc10, false, false);
        acc11 = __builtin_amdgcn_wmma_f32_16x16x32_f16(false, a1.v, false, b1.v, (short)0, acc11, false, false);
    }
    #pragma unroll
    for (int r = 0; r < 8; ++r) {
        #pragma unroll
        for (int ri = 0; ri < 2; ++ri) {
            int m = row0 + ri * 16 + r + half * 8;    // C/D layout: lanes 16-31 -> M += 8
            #pragma unroll
            for (int ci = 0; ci < 2; ++ci) {
                int n = col0 + ci * 16 + idx;
                float v = ri == 0 ? (ci == 0 ? acc00[r] : acc01[r])
                                  : (ci == 0 ? acc10[r] : acc11[r]);
                if (n < N) v += bias[n];
                if (relu) v = v > 0.0f ? v : 0.0f;
                if (C && n < N) C[(size_t)m * ldc + n] = v;
                if (C16 && n < Np16) C16[(size_t)m * ldc16 + n] = (_Float16)(n < N ? v : 0.0f);
            }
        }
    }
}

// ------------------------------------------------------------------
// Build ef16 = [ea | h[src,:48] | h[dst,:48]] (f16, K-padded to 160)
// ------------------------------------------------------------------
__global__ void k_ef(const float* __restrict__ ea, const float* __restrict__ h,
                     const int* __restrict__ src, const int* __restrict__ dst,
                     _Float16* __restrict__ ef16, int E) {
    int i = blockIdx.x * blockDim.x + threadIdx.x;
    if (i >= E * NEFP) return;
    int e = i / NEFP, c = i % NEFP;
    float v = 0.0f;
    if (c < 48)       v = ea[e * 48 + c];
    else if (c < 96)  v = h[src[e] * HPAD + (c - 48)];
    else if (c < NEF) v = h[dst[e] * HPAD + (c - 96)];
    ef16[i] = (_Float16)v;
}

// ------------------------------------------------------------------
// Equivariant tensor product for one edge chunk; accumulates sums by src.
// ------------------------------------------------------------------
__global__ void k_tp(TpMeta meta, const float* __restrict__ h, const float* __restrict__ esh,
                     const float* __restrict__ wbuf, int wn,
                     const int* __restrict__ src, const int* __restrict__ dst,
                     float* __restrict__ agg, const float* __restrict__ w3j,
                     int e0, int dim_in, int dim_out) {
    __shared__ float xs[HPAD];
    __shared__ float sh9[9];
    __shared__ float ob[HPAD];
    __shared__ float tmp[48 * 5];
    int le = blockIdx.x;
    int e = e0 + le;
    int t = threadIdx.x;
    const float* x1 = h + (size_t)dst[e] * HPAD;
    for (int c = t; c < HPAD; c += blockDim.x) { xs[c] = (c < dim_in) ? x1[c] : 0.0f; ob[c] = 0.0f; }
    if (t < 9) sh9[t] = esh[e * 9 + t];
    __syncthreads();
    for (int pi = 0; pi < meta.np; ++pi) {
        TpPath P = meta.p[pi];
        int d1 = 2 * P.l1 + 1, d2 = 2 * P.l2 + 1, d3 = 2 * P.l3 + 1;
        const float* C = w3j + (P.l1 * 9 + P.l2 * 3 + P.l3) * 125;
        int n1 = P.m1 * d3;
        for (int x = t; x < n1; x += blockDim.x) {
            int u = x / d3, k = x % d3;
            float s = 0.0f;
            for (int i = 0; i < d1; ++i) {
                float xv = xs[P.in_off + u * d1 + i];
                for (int j = 0; j < d2; ++j) s += xv * sh9[P.sh_off + j] * C[(i * d2 + j) * d3 + k];
            }
            tmp[x] = s;
        }
        __syncthreads();
        int n3 = P.m3 * d3;
        const float* wrow = wbuf + (size_t)le * wn + P.woff;
        for (int x = t; x < n3; x += blockDim.x) {
            int w = x / d3, k = x % d3;
            float s = 0.0f;
            for (int u = 0; u < P.m1; ++u) s += tmp[u * d3 + k] * wrow[u * P.m3 + w];
            ob[P.out_off + w * d3 + k] += P.coeff * s;
        }
        __syncthreads();
    }
    int sn = src[e];
    for (int c = t; c < dim_out; c += blockDim.x) atomicAdd(&agg[(size_t)sn * HPAD + c], ob[c]);
}

// ------------------------------------------------------------------
// upd = agg / max(cnt,1) + padded(h)
// ------------------------------------------------------------------
__global__ void k_upd(const float* __restrict__ agg, const float* __restrict__ cnt,
                      const float* __restrict__ hold, float* __restrict__ upd,
                      int N, int dim_out) {
    int i = blockIdx.x * blockDim.x + threadIdx.x;
    if (i >= N * HPAD) return;
    int n = i / HPAD, c = i % HPAD;
    if (c >= dim_out) return;
    float cc = cnt[n]; cc = cc > 1.0f ? cc : 1.0f;
    upd[i] = agg[i] / cc + hold[i];
}

// ------------------------------------------------------------------
// Per-channel irrep BN statistics (one block per channel)
// ------------------------------------------------------------------
__global__ void k_bnstats(IrrList irr, const float* __restrict__ upd,
                          const float* __restrict__ bn_w,
                          float* __restrict__ meanv, float* __restrict__ scalev, int N) {
    int ch = blockIdx.x;
    int ch0 = 0, col0 = 0, ii = 0;
    for (; ii < irr.n; ++ii) {
        int d = 2 * irr.l[ii] + 1;
        if (ch < ch0 + irr.m[ii]) break;
        ch0 += irr.m[ii]; col0 += irr.m[ii] * d;
    }
    int d = 2 * irr.l[ii] + 1;
    int col = col0 + (ch - ch0) * d;
    bool sc = (irr.l[ii] == 0 && irr.p[ii] == 1);
    __shared__ float ssum[256], ssq[256];
    float s = 0.0f, q = 0.0f;
    for (int x = threadIdx.x; x < N * d; x += blockDim.x) {
        int n = x / d, j = x % d;
        float v = upd[(size_t)n * HPAD + col + j];
        s += v; q += v * v;
    }
    ssum[threadIdx.x] = s; ssq[threadIdx.x] = q;
    __syncthreads();
    for (int st = 128; st > 0; st >>= 1) {
        if (threadIdx.x < st) { ssum[threadIdx.x] += ssum[threadIdx.x + st]; ssq[threadIdx.x] += ssq[threadIdx.x + st]; }
        __syncthreads();
    }
    if (threadIdx.x == 0) {
        float mean = sc ? ssum[0] / (float)N : 0.0f;
        float norm = sc ? (ssq[0] / (float)N - mean * mean) : (ssq[0] / (float)(N * d));
        meanv[ch] = mean;
        scalev[ch] = bn_w[ch] / sqrtf(norm + 1e-5f);
    }
}

// ------------------------------------------------------------------
// Apply BN + bias + relu + residual: h_new = padded(h_old) + relu(bn(upd))
// ------------------------------------------------------------------
__global__ void k_apply(IrrList irr, const float* __restrict__ upd,
                        const float* __restrict__ meanv, const float* __restrict__ scalev,
                        const float* __restrict__ bn_b, const float* __restrict__ hold,
                        float* __restrict__ hnew, int N, int dim_out) {
    int i = blockIdx.x * blockDim.x + threadIdx.x;
    if (i >= N * HPAD) return;
    int c = i % HPAD;
    if (c >= dim_out) { hnew[i] = 0.0f; return; }
    int ch0 = 0, col0 = 0, ii = 0;
    for (; ii < irr.n; ++ii) {
        int d = 2 * irr.l[ii] + 1, w = irr.m[ii] * d;
        if (c < col0 + w) break;
        ch0 += irr.m[ii]; col0 += w;
    }
    int d = 2 * irr.l[ii] + 1;
    int ch = ch0 + (c - col0) / d;
    bool sc = (irr.l[ii] == 0 && irr.p[ii] == 1);
    float v = (upd[i] - meanv[ch]) * scalev[ch];
    if (sc) v += bn_b[ch];     // scalar-even channels are always irrep 0 -> bias index == ch
    v = v > 0.0f ? v : 0.0f;
    hnew[i] = hold[i] + v;
}

// ------------------------------------------------------------------
// Readout: per-graph mean of relu(h[:,:48] @ out_W / sqrt(48))
// ------------------------------------------------------------------
__global__ void k_out(const float* __restrict__ h, const float* __restrict__ outW,
                      const int* __restrict__ batch, float* __restrict__ out, int N) {
    __shared__ float ss[256], sc[256];
    int g = blockIdx.x;
    float s = 0.0f, c = 0.0f;
    for (int n = threadIdx.x; n < N; n += blockDim.x) {
        if (batch[n] == g) {
            float dsum = 0.0f;
            for (int k = 0; k < NSC; ++k) dsum += h[(size_t)n * HPAD + k] * outW[k];
            dsum *= 0.14433756729740643f;     // 1/sqrt(48)
            dsum = dsum > 0.0f ? dsum : 0.0f;
            s += dsum; c += 1.0f;
        }
    }
    ss[threadIdx.x] = s; sc[threadIdx.x] = c;
    __syncthreads();
    for (int st = 128; st > 0; st >>= 1) {
        if (threadIdx.x < st) { ss[threadIdx.x] += ss[threadIdx.x + st]; sc[threadIdx.x] += sc[threadIdx.x + st]; }
        __syncthreads();
    }
    if (threadIdx.x == 0) out[g] = ss[0] / (sc[0] > 1.0f ? sc[0] : 1.0f);
}

// ==================================================================
// Host side
// ==================================================================
struct HIrr { int m, l, p; };

static int irr_dim_h(const HIrr* a, int n) {
    int s = 0; for (int i = 0; i < n; ++i) s += a[i].m * (2 * a[i].l + 1); return s;
}

static const HIrr SEQS[4][6] = {
    { {48,0,1}, {0,0,0}, {0,0,0}, {0,0,0}, {0,0,0}, {0,0,0} },
    { {48,0,1}, {10,1,-1}, {10,2,1}, {0,0,0}, {0,0,0}, {0,0,0} },
    { {48,0,1}, {10,1,-1}, {10,2,1}, {10,1,1}, {10,2,-1}, {0,0,0} },
    { {48,0,1}, {10,1,-1}, {10,2,1}, {10,1,1}, {10,2,-1}, {48,0,-1} },
};
static const int SEQN[4] = { 1, 3, 5, 6 };
static const HIrr SH_IRR_H[3] = { {1,0,1}, {1,1,-1}, {1,2,1} };
static const int SH_OFF_H[3] = { 0, 1, 4 };

static int build_tp_host(const HIrr* in, int nin, const HIrr* out, int nout, TpMeta* meta) {
    int fan[6] = { 0,0,0,0,0,0 };
    for (int i1 = 0; i1 < nin; ++i1)
      for (int i2 = 0; i2 < 3; ++i2)
        for (int i3 = 0; i3 < nout; ++i3) {
            int l1 = in[i1].l, l2 = SH_IRR_H[i2].l, l3 = out[i3].l;
            int lo = l1 - l2; if (lo < 0) lo = -lo;
            if (in[i1].p * SH_IRR_H[i2].p == out[i3].p && lo <= l3 && l3 <= l1 + l2)
                fan[i3] += in[i1].m;
        }
    int in_off[6], out_off[6];
    { int o = 0; for (int i = 0; i < nin; ++i) { in_off[i] = o; o += in[i].m * (2 * in[i].l + 1); } }
    { int o = 0; for (int i = 0; i < nout; ++i) { out_off[i] = o; o += out[i].m * (2 * out[i].l + 1); } }
    int woff = 0; meta->np = 0;
    for (int i1 = 0; i1 < nin; ++i1)
      for (int i2 = 0; i2 < 3; ++i2)
        for (int i3 = 0; i3 < nout; ++i3) {
            int l1 = in[i1].l, l2 = SH_IRR_H[i2].l, l3 = out[i3].l;
            int lo = l1 - l2; if (lo < 0) lo = -lo;
            if (!(in[i1].p * SH_IRR_H[i2].p == out[i3].p && lo <= l3 && l3 <= l1 + l2)) continue;
            TpPath& P = meta->p[meta->np++];
            P.in_off = in_off[i1]; P.m1 = in[i1].m; P.l1 = l1;
            P.sh_off = SH_OFF_H[i2]; P.l2 = l2;
            P.out_off = out_off[i3]; P.m3 = out[i3].m; P.l3 = l3;
            P.woff = woff;
            P.coeff = sqrtf((float)(2 * l3 + 1) / (float)fan[i3]);
            woff += in[i1].m * out[i3].m;
        }
    return woff;   // = wn
}

extern "C" void kernel_launch(void* const* d_in, const int* in_sizes, int n_in,
                              void* d_out, int out_size, void* d_ws, size_t ws_size,
                              hipStream_t stream) {
    const int N = in_sizes[0];
    const int E = in_sizes[3] / 2;
    const int G = out_size;

    const int*   x_cat    = (const int*)d_in[0];
    const float* x_scalar = (const float*)d_in[1];
    const float* pos      = (const float*)d_in[2];
    const int*   src      = (const int*)d_in[3];
    const int*   dst      = (const int*)d_in[3] + E;
    const float* eattr_in = (const float*)d_in[4];
    const int*   batch    = (const int*)d_in[5];
    // d_in[6] = n_graphs scalar (use out_size instead)
    const float* emb_tab  = (const float*)d_in[7];
    const float* node_W   = (const float*)d_in[8];
    const float* node_b   = (const float*)d_in[9];
    const float* edge_W1  = (const float*)d_in[10];
    const float* edge_b1  = (const float*)d_in[11];
    const float* edge_W2  = (const float*)d_in[12];
    const float* edge_b2  = (const float*)d_in[13];
    const float* out_W    = (const float*)d_in[14];

    // ---- workspace layout (byte cursor, 256B aligned carves) ----
    char* base = (char*)d_ws;
    size_t o = 0;
    auto carve = [&](size_t bytes) -> char* {
        char* p = base + o;
        o += (bytes + 255) & ~(size_t)255;
        return p;
    };
    float*    w3j_tab  = (float*)carve(27 * 125 * 4);
    float*    h_a      = (float*)carve((size_t)N * HPAD * 4);
    float*    h_b      = (float*)carve((size_t)N * HPAD * 4);
    float*    esh      = (float*)carve((size_t)E * 9 * 4);
    float*    ea       = (float*)carve((size_t)E * NSC * 4);
    float*    agg      = (float*)carve((size_t)N * HPAD * 4);
    float*    upd      = (float*)carve((size_t)N * HPAD * 4);
    float*    cnt      = (float*)carve((size_t)N * 4);
    float*    meanv    = (float*)carve(160 * 4);
    float*    scalev   = (float*)carve(160 * 4);
    _Float16* eattr16  = (_Float16*)carve((size_t)E * EATTRP * 2);
    _Float16* hid16    = (_Float16*)carve((size_t)E * EATTRP * 2);
    _Float16* ef16     = (_Float16*)carve((size_t)E * NEFP * 2);
    _Float16* fc1o16   = (_Float16*)carve((size_t)E * NEFP * 2);
    _Float16* w1T      = (_Float16*)carve((size_t)64 * 64 * 2);
    _Float16* w2T      = (_Float16*)carve((size_t)64 * 64 * 2);
    _Float16* fc1T     = (_Float16*)carve((size_t)192 * NEFP * 2);
    _Float16* fc2T     = (_Float16*)carve((size_t)10496 * NEFP * 2);
    float*    wchunk   = (float*)carve((size_t)CHUNK * 10448 * 4);
    (void)ws_size;

    // ---- precompute / init ----
    hipLaunchKernelGGL(k_w3j, dim3(1), dim3(32), 0, stream, w3j_tab);
    hipLaunchKernelGGL(k_node_init, dim3((N * HPAD + 255) / 256), dim3(256), 0, stream,
                       x_cat, x_scalar, emb_tab, node_W, node_b, h_a, N);
    hipLaunchKernelGGL(k_edge_feat, dim3((E + 63) / 64), dim3(64), 0, stream,
                       pos, src, dst, eattr_in, eattr16, esh, E);
    hipLaunchKernelGGL(k_zero, dim3((N + 255) / 256), dim3(256), 0, stream, cnt, N);
    hipLaunchKernelGGL(k_cnt, dim3((E + 255) / 256), dim3(256), 0, stream, src, cnt, E);

    // ---- edge MLP: ea = relu(eattr@W1+b1)@W2+b2 ----
    hipLaunchKernelGGL(k_cvtBT, dim3((64 * 64 + 255) / 256), dim3(256), 0, stream,
                       edge_W1, EATTRW, NSC, w1T, EATTRP, 64);
    hipLaunchKernelGGL(k_cvtBT, dim3((64 * 64 + 255) / 256), dim3(256), 0, stream,
                       edge_W2, NSC, NSC, w2T, EATTRP, 64);
    // hid16 = relu(eattr16 @ W1 + b1)  (f16-out only)
    hipLaunchKernelGGL(k_gemm16, dim3(1, E / 64), dim3(128), 0, stream,
                       eattr16, EATTRP, w1T, EATTRP, edge_b1,
                       (float*)nullptr, 0, hid16, EATTRP,
                       NSC, EATTRP, EATTRP, 1);
    // ea = hid16 @ W2 + b2  (f32-out only)
    hipLaunchKernelGGL(k_gemm16, dim3(1, E / 64), dim3(128), 0, stream,
                       hid16, EATTRP, w2T, EATTRP, edge_b2,
                       ea, NSC, (_Float16*)nullptr, 0,
                       NSC, EATTRP, 0, 0);

    // ---- layers ----
    float* cur = h_a;
    float* nxt = h_b;
    for (int li = 0; li < 6; ++li) {
        int si = li < 3 ? li : 3;
        int so = (li + 1) < 3 ? (li + 1) : 3;
        const HIrr* IN = SEQS[si];  int nin = SEQN[si];
        const HIrr* OUT = SEQS[so]; int nout = SEQN[so];
        int dim_in = irr_dim_h(IN, nin);
        int dim_out = irr_dim_h(OUT, nout);
        TpMeta meta;
        int wn = build_tp_host(IN, nin, OUT, nout, &meta);
        int npad = ((wn + 63) / 64) * 64;
        int nmul = 0; for (int i = 0; i < nout; ++i) nmul += OUT[i].m;
        IrrList irr; irr.n = nout;
        for (int i = 0; i < nout; ++i) { irr.m[i] = OUT[i].m; irr.l[i] = OUT[i].l; irr.p[i] = OUT[i].p; }

        const float* fc1_W = (const float*)d_in[15 + 6 * li];
        const float* fc1_b = (const float*)d_in[16 + 6 * li];
        const float* fc2_W = (const float*)d_in[17 + 6 * li];
        const float* fc2_b = (const float*)d_in[18 + 6 * li];
        const float* bn_w  = (const float*)d_in[19 + 6 * li];
        const float* bn_b  = (const float*)d_in[20 + 6 * li];

        // weight conversions (f32 -> f16 transposed, zero padded)
        hipLaunchKernelGGL(k_cvtBT, dim3((192 * NEFP + 255) / 256), dim3(256), 0, stream,
                           fc1_W, NEF, NEF, fc1T, NEFP, 192);
        hipLaunchKernelGGL(k_cvtBT, dim3((npad * NEFP + 255) / 256), dim3(256), 0, stream,
                           fc2_W, NEF, wn, fc2T, NEFP, npad);

        // ef16 = [ea | h[src,:48] | h[dst,:48]] (padded f16)
        hipLaunchKernelGGL(k_ef, dim3((E * NEFP + 255) / 256), dim3(256), 0, stream,
                           ea, cur, src, dst, ef16, E);
        // fc1o16 = relu(ef16 @ fc1_W + b)
        hipLaunchKernelGGL(k_gemm16, dim3(3, E / 64), dim3(128), 0, stream,
                           ef16, NEFP, fc1T, NEFP, fc1_b,
                           (float*)nullptr, 0, fc1o16, NEFP,
                           NEF, NEFP, NEFP, 1);
        // zero segment accumulator
        hipLaunchKernelGGL(k_zero, dim3((N * HPAD + 255) / 256), dim3(256), 0, stream, agg, N * HPAD);
        // chunked fc2 + tensor-product (keeps weight tile L2-resident)
        for (int c0 = 0; c0 < E; c0 += CHUNK) {
            int mrows = (E - c0) < CHUNK ? (E - c0) : CHUNK;
            hipLaunchKernelGGL(k_gemm16, dim3(npad / 64, mrows / 64), dim3(128), 0, stream,
                               fc1o16 + (size_t)c0 * NEFP, NEFP, fc2T, NEFP, fc2_b,
                               wchunk, wn, (_Float16*)nullptr, 0,
                               wn, NEFP, 0, 0);
            hipLaunchKernelGGL(k_tp, dim3(mrows), dim3(64), 0, stream,
                               meta, cur, esh, wchunk, wn, src, dst, agg, w3j_tab,
                               c0, dim_in, dim_out);
        }
        // upd = agg/cnt + pad(h)
        hipLaunchKernelGGL(k_upd, dim3((N * HPAD + 255) / 256), dim3(256), 0, stream,
                           agg, cnt, cur, upd, N, dim_out);
        // batch norm over irreps
        hipLaunchKernelGGL(k_bnstats, dim3(nmul), dim3(256), 0, stream,
                           irr, upd, bn_w, meanv, scalev, N);
        hipLaunchKernelGGL(k_apply, dim3((N * HPAD + 255) / 256), dim3(256), 0, stream,
                           irr, upd, meanv, scalev, bn_b, cur, nxt, N, dim_out);
        float* t = cur; cur = nxt; nxt = t;
    }

    // ---- readout ----
    hipLaunchKernelGGL(k_out, dim3(G), dim3(256), 0, stream, cur, out_W, batch, (float*)d_out, N);
}
